// TemporalFusionLayer_26164940767821
// MI455X (gfx1250) — matrix operand order repaired and checked by reading, not compiled, not run on hardware
//
#include <hip/hip_runtime.h>
#include <cstdint>
#include <cstddef>

// ---------------- problem constants ----------------
#define BB 2
#define TT 683
#define LL 2049
#define DD 1024
#define HH 16
#define MLPD 4096
#define MROWS 4098          // B*L
#define QKVLD 3072          // 3*D
#define KTILES 129          // ceil(L/16)
#define VTLD 2080           // padded key length for V^T (>= 65*32)
#define MLLD 2064           // padded rows for m/l stats and LDS attn tile cols

typedef _Float16 half8 __attribute__((ext_vector_type(8)));
typedef _Float16 v16h  __attribute__((ext_vector_type(16)));
typedef float    v8f   __attribute__((ext_vector_type(8)));

__device__ __forceinline__ v8f zero8() {
  v8f z;
#pragma unroll
  for (int i = 0; i < 8; ++i) z[i] = 0.f;
  return z;
}
__device__ __forceinline__ v16h zero16h() {
  v16h z;
#pragma unroll
  for (int i = 0; i < 16; ++i) z[i] = (_Float16)0.f;
  return z;
}

__device__ __forceinline__ v16h pack16(half8 c0, half8 c1) {
  v16h out;
#pragma unroll
  for (int i = 0; i < 8; ++i) { out[i] = c0[i]; out[8 + i] = c1[i]; }
  return out;
}

// A-fragment 16x32 f16 (MxK): lane(0-15)=row M (lane&15), K chunks
// [hi*8, hi*8+8) -> VGPR0..3 and [16+hi*8, ...) -> VGPR4..7 (CDNA5 §7.12.2)
__device__ __forceinline__ v16h load_frag_a(const _Float16* base, int ld, int row_limit,
                                            int row0, int col0) {
  int lane = threadIdx.x & 31;
  int m = lane & 15, hi = lane >> 4;
  const _Float16* p = base + (size_t)(row0 + m) * (size_t)ld + col0 + hi * 8;
  bool ok = (row0 + 16 <= row_limit) | (row0 + m < row_limit);
  if (ok)
    return pack16(*(const half8*)p, *(const half8*)(p + 16));
  return zero16h();
}

// B-fragment 32x16 f16 (KxN) loaded from "W rows" (W[n][k] row-major, n = output col):
// lane(0-15)=col N, K chunks [hi*8, hi*8+8) and [16+hi*8, 16+hi*8+8) (same order as A)
__device__ __forceinline__ v16h load_frag_b(const _Float16* base, int ld, int row_limit,
                                            int row0, int col0) {
  int lane = threadIdx.x & 31;
  int c = lane & 15, hi = lane >> 4;
  const _Float16* p = base + (size_t)(row0 + c) * (size_t)ld + col0 + hi * 8;
  bool ok = (row0 + 16 <= row_limit) | (row0 + c < row_limit);
  if (ok)
    return pack16(*(const half8*)p, *(const half8*)(p + 16));
  return zero16h();
}

__device__ __forceinline__ v8f wmma_f16(v16h a, v16h b, v8f c) {
  return __builtin_amdgcn_wmma_f32_16x16x32_f16(false, a, false, b, (short)0, c,
                                                false, false);
}

// ---- CDNA5 async global->LDS copy (ASYNCcnt path, §CDNA5 ISA) ----
// LDS generic addresses carry the LDS byte offset in addr[31:0].
__device__ __forceinline__ void async_g2l_b128(const void* lds_ptr, const void* gptr) {
  uint32_t lds_off = (uint32_t)(uintptr_t)lds_ptr;
  asm volatile("global_load_async_to_lds_b128 %0, %1, off"
               :: "v"(lds_off), "v"(gptr) : "memory");
}
__device__ __forceinline__ void wait_async0() {
  asm volatile("s_wait_asynccnt 0x0" ::: "memory");
}

// ---------------- elementwise convert f32 -> f16 ----------------
__global__ void cvt_f16_kernel(const float* __restrict__ src, _Float16* __restrict__ dst,
                               size_t n) {
  size_t i = (size_t)blockIdx.x * 256 + threadIdx.x;
  if (i < n) dst[i] = (_Float16)src[i];
}

// ---------------- generic GEMM: out = act(A[M,K] @ W[N,K]^T + bias) ----------------
// block = 256 threads (8 waves); each wave does a 16x64 tile, all waves share the
// 64-col W panel. W k-tiles (64x32 f16 = 4KB) are double-buffered in LDS via the
// async global->LDS engine so the DMA overlaps the WMMAs. B-fragments are loaded
// into distinct registers so the ds_loads clause and the 4 WMMAs issue back-to-back.
__global__ __launch_bounds__(256) void gemm_f16_kernel(
    const _Float16* __restrict__ A, const _Float16* __restrict__ W,
    const float* __restrict__ bias, _Float16* __restrict__ out16,
    float* __restrict__ out32, int M, int N, int K, int act) {
  __shared__ _Float16 wtile[2][64 * 32];
  int lane = threadIdx.x & 31, w = threadIdx.x >> 5;
  int row0 = blockIdx.x * 128 + w * 16;
  int col0 = blockIdx.y * 64;
  // staging assignment: thread t copies 16B: row t>>2 of the panel, chunk t&3
  int srow = (int)threadIdx.x >> 2, schunk = (int)threadIdx.x & 3;
  const _Float16* wsrc = W + (size_t)(col0 + srow) * (size_t)K + schunk * 8;
  _Float16* ldst0 = &wtile[0][srow * 32 + schunk * 8];
  _Float16* ldst1 = &wtile[1][srow * 32 + schunk * 8];

  v8f acc[4];
#pragma unroll
  for (int g = 0; g < 4; ++g) acc[g] = zero8();

  const int nk = K / 32;
  async_g2l_b128(ldst0, wsrc);              // prologue: stage k-tile 0 into buf 0
  for (int ki = 0; ki < nk; ++ki) {
    wait_async0();                          // this wave's staged tile has landed
    __syncthreads();                        // all waves' stages have landed
    if (ki + 1 < nk)                        // stage next tile into the other buffer
      async_g2l_b128((ki & 1) ? ldst0 : ldst1, wsrc + (size_t)(ki + 1) * 32);
    if (ki + 2 < nk)                        // prefetch A stream (global_prefetch_b8)
      __builtin_prefetch(A + (size_t)(row0 + ((int)threadIdx.x & 15)) * (size_t)K +
                             (size_t)(ki + 2) * 32, 0, 0);
    const _Float16* wb = wtile[ki & 1];
    v16h a  = load_frag_a(A, K, M, row0, ki * 32);
    v16h b0 = load_frag_b(wb, 32, 1 << 30, 0, 0);
    v16h b1 = load_frag_b(wb, 32, 1 << 30, 16, 0);
    v16h b2 = load_frag_b(wb, 32, 1 << 30, 32, 0);
    v16h b3 = load_frag_b(wb, 32, 1 << 30, 48, 0);
    acc[0] = wmma_f16(a, b0, acc[0]);
    acc[1] = wmma_f16(a, b1, acc[1]);
    acc[2] = wmma_f16(a, b2, acc[2]);
    acc[3] = wmma_f16(a, b3, acc[3]);
    __syncthreads();                        // reads done before buffer is re-staged
  }

  int m = lane & 15, hi = lane >> 4;
#pragma unroll
  for (int g = 0; g < 4; ++g) {
    int col = col0 + g * 16 + m;
    float bv = bias ? bias[col] : 0.f;
#pragma unroll
    for (int r = 0; r < 8; ++r) {
      int row = row0 + r + hi * 8;
      if (row < M) {
        float v = acc[g][r] + bv;
        if (act) v = 0.5f * v * (1.f + erff(v * 0.70710678118f));  // exact GELU
        size_t o = (size_t)row * (size_t)N + col;
        if (out32) out32[o] = v;
        if (out16) out16[o] = (_Float16)v;
      }
    }
  }
}

// ---------------- V^T build: vt[b][h][d][col] = V[b,col,h,d], zero padded ----------------
__global__ void vt_build_kernel(const _Float16* __restrict__ qkv, _Float16* __restrict__ vt) {
  size_t idx = (size_t)blockIdx.x * 256 + threadIdx.x;
  size_t total = (size_t)BB * HH * 64 * VTLD;
  if (idx >= total) return;
  int col = (int)(idx % VTLD);
  int d = (int)((idx / VTLD) % 64);
  int h = (int)((idx / ((size_t)VTLD * 64)) % HH);
  int b = (int)(idx / ((size_t)VTLD * 64 * HH));
  _Float16 val = (_Float16)0.f;
  if (col < LL)
    val = qkv[((size_t)b * LL + col) * QKVLD + 2048 + h * 64 + d];
  vt[idx] = val;
}

// ---------------- attention pass 1: per-row max / sumexp ----------------
__global__ __launch_bounds__(256) void attn_pass1_kernel(
    const _Float16* __restrict__ qkv, float* __restrict__ mbuf, float* __restrict__ lbuf) {
  int lane = threadIdx.x & 31, w = threadIdx.x >> 5;
  int m_ = lane & 15, hi = lane >> 4;
  int tile = blockIdx.x, h = blockIdx.y, b = blockIdx.z;
  int i0 = tile * 16;
  const _Float16* qbase = qkv + (size_t)b * LL * QKVLD + h * 64;
  const _Float16* kbase = qbase + 1024;
  v16h qa0 = load_frag_a(qbase, QKVLD, LL, i0, 0);
  v16h qa1 = load_frag_a(qbase, QKVLD, LL, i0, 32);
  float mrun[8], lrun[8];
#pragma unroll
  for (int r = 0; r < 8; ++r) { mrun[r] = -1e30f; lrun[r] = 0.f; }
  for (int j = w; j < KTILES; j += 8) {
    v16h kb0 = load_frag_b(kbase, QKVLD, LL, j * 16, 0);
    v16h kb1 = load_frag_b(kbase, QKVLD, LL, j * 16, 32);
    v8f s = zero8();
    s = wmma_f16(qa0, kb0, s);
    s = wmma_f16(qa1, kb1, s);
    int col = j * 16 + m_;
    bool cv = col < LL;
#pragma unroll
    for (int r = 0; r < 8; ++r) {
      float sv = cv ? s[r] * 0.125f : -3.0e38f;
      float mn = fmaxf(mrun[r], sv);
      lrun[r] = lrun[r] * __expf(mrun[r] - mn) + __expf(sv - mn);
      mrun[r] = mn;
    }
  }
  // reduce across the 16 lanes (columns) sharing each row
#pragma unroll
  for (int mask = 1; mask < 16; mask <<= 1) {
#pragma unroll
    for (int r = 0; r < 8; ++r) {
      float om = __shfl_xor(mrun[r], mask, 32);
      float ol = __shfl_xor(lrun[r], mask, 32);
      float mn = fmaxf(mrun[r], om);
      lrun[r] = lrun[r] * __expf(mrun[r] - mn) + ol * __expf(om - mn);
      mrun[r] = mn;
    }
  }
  __shared__ float mls[8][16], lls[8][16];
  if (m_ == 0) {
#pragma unroll
    for (int r = 0; r < 8; ++r) { mls[w][r + hi * 8] = mrun[r]; lls[w][r + hi * 8] = lrun[r]; }
  }
  __syncthreads();
  if (threadIdx.x < 16) {
    float mm = -3.0e38f, ll = 0.f;
    for (int ww = 0; ww < 8; ++ww) {
      float om = mls[ww][threadIdx.x], ol = lls[ww][threadIdx.x];
      float mn = fmaxf(mm, om);
      ll = ll * __expf(mm - mn) + ol * __expf(om - mn);
      mm = mn;
    }
    size_t o = (size_t)(b * HH + h) * MLLD + i0 + threadIdx.x;
    mbuf[o] = mm;
    lbuf[o] = ll;
  }
}

// ---------------- attention pass 2: probs, P@V, mean-attn accumulation ----------------
// grid (KTILES, B); loops all 16 heads; LDS holds attn-mean tile [16][MLLD].
__global__ __launch_bounds__(256) void attn_pass2_kernel(
    const _Float16* __restrict__ qkv, const _Float16* __restrict__ vt,
    const float* __restrict__ mbuf, const float* __restrict__ lbuf,
    _Float16* __restrict__ sa16, float* __restrict__ attn_out) {
  extern __shared__ char smem[];
  float* attnS = (float*)smem;                       // 16*MLLD f32
  float* Opart = attnS + 16 * MLLD;                  // 8*16*64 f32
  _Float16* plds = (_Float16*)(Opart + 8 * 16 * 64); // 8*16*32 f16
  int lane = threadIdx.x & 31, w = threadIdx.x >> 5;
  int m_ = lane & 15, hi = lane >> 4;
  int tile = blockIdx.x, b = blockIdx.y;
  int i0 = tile * 16;
  for (int i = threadIdx.x; i < 16 * MLLD; i += 256) attnS[i] = 0.f;
  __syncthreads();
  _Float16* pw = plds + w * 16 * 32;

  for (int h = 0; h < HH; ++h) {
    const _Float16* qbase = qkv + (size_t)b * LL * QKVLD + h * 64;
    const _Float16* kbase = qbase + 1024;
    const _Float16* vbase = vt + (size_t)(b * HH + h) * 64 * VTLD;
    v16h qa0 = load_frag_a(qbase, QKVLD, LL, i0, 0);
    v16h qa1 = load_frag_a(qbase, QKVLD, LL, i0, 32);
    float mf[8], lf[8];
#pragma unroll
    for (int r = 0; r < 8; ++r) {
      size_t o = (size_t)(b * HH + h) * MLLD + i0 + r + hi * 8;
      mf[r] = mbuf[o];
      lf[r] = lbuf[o];
    }
    v8f o4[4];
#pragma unroll
    for (int g = 0; g < 4; ++g) o4[g] = zero8();

    for (int jp = w; jp < 65; jp += 8) {           // pairs of 16-wide key tiles
#pragma unroll
      for (int t2 = 0; t2 < 2; ++t2) {
        int j = jp * 2 + t2;
        float pr[8];
        if (j < KTILES) {
          v16h kb0 = load_frag_b(kbase, QKVLD, LL, j * 16, 0);
          v16h kb1 = load_frag_b(kbase, QKVLD, LL, j * 16, 32);
          v8f s = zero8();
          s = wmma_f16(qa0, kb0, s);
          s = wmma_f16(qa1, kb1, s);
          int col = j * 16 + m_;
          bool cv = col < LL;
#pragma unroll
          for (int r = 0; r < 8; ++r) {
            pr[r] = cv ? __expf(s[r] * 0.125f - mf[r]) / lf[r] : 0.f;
            attnS[(r + hi * 8) * MLLD + col] += pr[r];   // unique owner thread
          }
        } else {
#pragma unroll
          for (int r = 0; r < 8; ++r) pr[r] = 0.f;
        }
#pragma unroll
        for (int r = 0; r < 8; ++r)
          pw[(r + hi * 8) * 32 + t2 * 16 + m_] = (_Float16)pr[r];
      }
      // P (16x32, staged row-major in per-wave LDS) @ V (32x64 via V^T rows)
      v16h pa  = load_frag_a(pw, 32, 1 << 30, 0, 0);
      v16h vb0 = load_frag_b(vbase, VTLD, 1 << 30, 0, jp * 32);
      v16h vb1 = load_frag_b(vbase, VTLD, 1 << 30, 16, jp * 32);
      v16h vb2 = load_frag_b(vbase, VTLD, 1 << 30, 32, jp * 32);
      v16h vb3 = load_frag_b(vbase, VTLD, 1 << 30, 48, jp * 32);
      o4[0] = wmma_f16(pa, vb0, o4[0]);
      o4[1] = wmma_f16(pa, vb1, o4[1]);
      o4[2] = wmma_f16(pa, vb2, o4[2]);
      o4[3] = wmma_f16(pa, vb3, o4[3]);
    }
    // reduce partial O over the 8 waves
#pragma unroll
    for (int g = 0; g < 4; ++g)
#pragma unroll
      for (int r = 0; r < 8; ++r)
        Opart[(w * 16 + r + hi * 8) * 64 + g * 16 + m_] = o4[g][r];
    __syncthreads();
    for (int e = threadIdx.x; e < 16 * 64; e += 256) {
      int rl = e >> 6, d = e & 63;
      float sum = 0.f;
      for (int ww = 0; ww < 8; ++ww) sum += Opart[(ww * 16 + rl) * 64 + d];
      int row = i0 + rl;
      if (row < LL)
        sa16[((size_t)b * LL + row) * DD + h * 64 + d] = (_Float16)sum;
    }
    __syncthreads();
  }
  // write head-averaged attention weights
  for (int e = threadIdx.x; e < 16 * MLLD; e += 256) {
    int rl = e / MLLD, col = e % MLLD;
    int row = i0 + rl;
    if (row < LL && col < LL)
      attn_out[((size_t)b * LL + row) * LL + col] = attnS[e] * 0.0625f;
  }
}

// ---------------- residual + LayerNorm ----------------
__global__ __launch_bounds__(256) void ln_residual_kernel(
    const float* __restrict__ xprev, const float* __restrict__ delta,
    const float* __restrict__ g, const float* __restrict__ be,
    float* __restrict__ outf, _Float16* __restrict__ outh) {
  int row = blockIdx.x;
  const float* xp = xprev + (size_t)row * DD;
  const float* dl = delta + (size_t)row * DD;
  float v[4], s = 0.f, sq = 0.f;
#pragma unroll
  for (int i = 0; i < 4; ++i) {
    int c = threadIdx.x + i * 256;
    float t = xp[c] + dl[c];
    v[i] = t; s += t; sq += t * t;
  }
#pragma unroll
  for (int mask = 1; mask < 32; mask <<= 1) {
    s += __shfl_xor(s, mask, 32);
    sq += __shfl_xor(sq, mask, 32);
  }
  __shared__ float ss[8], qq[8];
  int w = threadIdx.x >> 5, lane = threadIdx.x & 31;
  if (lane == 0) { ss[w] = s; qq[w] = sq; }
  __syncthreads();
  if (threadIdx.x == 0) {
    float a = 0.f, b2 = 0.f;
    for (int i = 0; i < 8; ++i) { a += ss[i]; b2 += qq[i]; }
    ss[0] = a; qq[0] = b2;
  }
  __syncthreads();
  float mean = ss[0] * (1.f / DD);
  float var = qq[0] * (1.f / DD) - mean * mean;
  float inv = rsqrtf(var + 1e-5f);
#pragma unroll
  for (int i = 0; i < 4; ++i) {
    int c = threadIdx.x + i * 256;
    float y = (v[i] - mean) * inv * g[c] + be[c];
    size_t o = (size_t)row * DD + c;
    outf[o] = y;
    if (outh) outh[o] = (_Float16)y;
  }
}

// ---------------- tiny per-timestep 3x3 cross-attention ----------------
__global__ void cross_attn_kernel(const _Float16* __restrict__ qkv2,
                                  _Float16* __restrict__ ca16) {
  int idx = blockIdx.x * 256 + threadIdx.x;
  const int total = BB * TT * 3 * HH;
  if (idx >= total) return;
  int hh = idx & 15;
  int q = (idx >> 4) % 3;
  int t = (idx / (16 * 3)) % TT;
  int b = idx / (16 * 3 * TT);
  size_t m0 = (size_t)b * LL + (size_t)t * 3;
  const _Float16* Qp = qkv2 + (m0 + q) * QKVLD + hh * 64;
  float s[3];
#pragma unroll
  for (int k = 0; k < 3; ++k) {
    const _Float16* Kp = qkv2 + (m0 + k) * QKVLD + 1024 + hh * 64;
    float acc = 0.f;
    for (int d = 0; d < 64; ++d) acc += (float)Qp[d] * (float)Kp[d];
    s[k] = acc * 0.125f;
  }
  float mx = fmaxf(s[0], fmaxf(s[1], s[2]));
  float e0 = __expf(s[0] - mx), e1 = __expf(s[1] - mx), e2 = __expf(s[2] - mx);
  float inv = 1.f / (e0 + e1 + e2);
  const _Float16* V0 = qkv2 + (m0 + 0) * QKVLD + 2048 + hh * 64;
  const _Float16* V1 = qkv2 + (m0 + 1) * QKVLD + 2048 + hh * 64;
  const _Float16* V2 = qkv2 + (m0 + 2) * QKVLD + 2048 + hh * 64;
  _Float16* op = ca16 + (m0 + q) * DD + hh * 64;
  for (int d = 0; d < 64; ++d) {
    float o = (e0 * (float)V0[d] + e1 * (float)V1[d] + e2 * (float)V2[d]) * inv;
    op[d] = (_Float16)o;
  }
}

// ---------------- host orchestration ----------------
static inline size_t alignup(size_t x) { return (x + 255) & ~(size_t)255; }

extern "C" void kernel_launch(void* const* d_in, const int* in_sizes, int n_in,
                              void* d_out, int out_size, void* d_ws, size_t ws_size,
                              hipStream_t stream) {
  (void)in_sizes; (void)n_in; (void)out_size; (void)ws_size;
  const float* x       = (const float*)d_in[0];
  const float* sa_in_w = (const float*)d_in[1];
  const float* sa_in_b = (const float*)d_in[2];
  const float* sa_out_w= (const float*)d_in[3];
  const float* sa_out_b= (const float*)d_in[4];
  const float* ca_in_w = (const float*)d_in[5];
  const float* ca_in_b = (const float*)d_in[6];
  const float* ca_out_w= (const float*)d_in[7];
  const float* ca_out_b= (const float*)d_in[8];
  const float* w1      = (const float*)d_in[9];
  const float* b1      = (const float*)d_in[10];
  const float* w2      = (const float*)d_in[11];
  const float* b2      = (const float*)d_in[12];
  const float* g1 = (const float*)d_in[13]; const float* be1 = (const float*)d_in[14];
  const float* g2 = (const float*)d_in[15]; const float* be2 = (const float*)d_in[16];
  const float* g3 = (const float*)d_in[17]; const float* be3 = (const float*)d_in[18];

  char* ws = (char*)d_ws;
  size_t off = 0;
  auto take = [&](size_t bytes) { char* p = ws + off; off = alignup(off + bytes); return p; };

  _Float16* x16        = (_Float16*)take((size_t)MROWS * DD * 2);
  _Float16* sa_in_w16  = (_Float16*)take((size_t)QKVLD * DD * 2);
  _Float16* sa_out_w16 = (_Float16*)take((size_t)DD * DD * 2);
  _Float16* ca_in_w16  = (_Float16*)take((size_t)QKVLD * DD * 2);
  _Float16* ca_out_w16 = (_Float16*)take((size_t)DD * DD * 2);
  _Float16* w1_16      = (_Float16*)take((size_t)MLPD * DD * 2);
  _Float16* w2_16      = (_Float16*)take((size_t)DD * MLPD * 2);
  _Float16* qkv16      = (_Float16*)take((size_t)MROWS * QKVLD * 2);
  _Float16* vt16       = (_Float16*)take((size_t)BB * HH * 64 * VTLD * 2);
  float*    mbuf       = (float*)take((size_t)BB * HH * MLLD * 4);
  float*    lbuf       = (float*)take((size_t)BB * HH * MLLD * 4);
  _Float16* act16      = (_Float16*)take((size_t)MROWS * DD * 2);   // sa16 / ca16
  float*    gout       = (float*)take((size_t)MROWS * DD * 4);
  float*    xcur       = (float*)take((size_t)MROWS * DD * 4);
  _Float16* xcur16     = (_Float16*)take((size_t)MROWS * DD * 2);
  _Float16* h16        = (_Float16*)take((size_t)MROWS * MLPD * 2);

  float* out_x    = (float*)d_out;
  float* out_attn = out_x + (size_t)MROWS * DD;

  auto cvt = [&](const float* s, _Float16* d, size_t n) {
    cvt_f16_kernel<<<(unsigned)((n + 255) / 256), 256, 0, stream>>>(s, d, n);
  };
  cvt(x, x16, (size_t)MROWS * DD);
  cvt(sa_in_w, sa_in_w16, (size_t)QKVLD * DD);
  cvt(sa_out_w, sa_out_w16, (size_t)DD * DD);
  cvt(ca_in_w, ca_in_w16, (size_t)QKVLD * DD);
  cvt(ca_out_w, ca_out_w16, (size_t)DD * DD);
  cvt(w1, w1_16, (size_t)MLPD * DD);
  cvt(w2, w2_16, (size_t)DD * MLPD);

  const unsigned mblk = (MROWS + 127) / 128;   // 33
  // QKV projection (self-attn)
  gemm_f16_kernel<<<dim3(mblk, QKVLD / 64), 256, 0, stream>>>(
      x16, sa_in_w16, sa_in_b, qkv16, nullptr, MROWS, QKVLD, DD, 0);
  // V^T
  {
    size_t n = (size_t)BB * HH * 64 * VTLD;
    vt_build_kernel<<<(unsigned)((n + 255) / 256), 256, 0, stream>>>(qkv16, vt16);
  }
  // attention passes
  attn_pass1_kernel<<<dim3(KTILES, HH, BB), 256, 0, stream>>>(qkv16, mbuf, lbuf);
  {
    size_t smem = (size_t)(16 * MLLD + 8 * 16 * 64) * 4 + (size_t)(8 * 16 * 32) * 2;
    attn_pass2_kernel<<<dim3(KTILES, BB), 256, smem, stream>>>(qkv16, vt16, mbuf, lbuf,
                                                               act16, out_attn);
  }
  // out projection + residual LN1
  gemm_f16_kernel<<<dim3(mblk, DD / 64), 256, 0, stream>>>(
      act16, sa_out_w16, sa_out_b, nullptr, gout, MROWS, DD, DD, 0);
  ln_residual_kernel<<<MROWS, 256, 0, stream>>>(x, gout, g1, be1, xcur, xcur16);

  // cross-attention projections (reuse qkv16)
  gemm_f16_kernel<<<dim3(mblk, QKVLD / 64), 256, 0, stream>>>(
      xcur16, ca_in_w16, ca_in_b, qkv16, nullptr, MROWS, QKVLD, DD, 0);
  {
    int total = BB * TT * 3 * HH;
    cross_attn_kernel<<<(total + 255) / 256, 256, 0, stream>>>(qkv16, act16);
  }
  gemm_f16_kernel<<<dim3(mblk, DD / 64), 256, 0, stream>>>(
      act16, ca_out_w16, ca_out_b, nullptr, gout, MROWS, DD, DD, 0);
  ln_residual_kernel<<<MROWS, 256, 0, stream>>>(xcur, gout, g2, be2, xcur, xcur16);

  // FFN
  gemm_f16_kernel<<<dim3(mblk, MLPD / 64), 256, 0, stream>>>(
      xcur16, w1_16, b1, h16, nullptr, MROWS, MLPD, DD, 1 /*gelu*/);
  gemm_f16_kernel<<<dim3(mblk, DD / 64), 256, 0, stream>>>(
      h16, w2_16, b2, nullptr, gout, MROWS, DD, MLPD, 0);
  ln_residual_kernel<<<MROWS, 256, 0, stream>>>(xcur, gout, g3, be3, out_x, nullptr);
}
